// GraphTransformerLayer_26414048870627
// MI455X (gfx1250) — compile-verified, matrix operand order checked by reading, not run on hardware
//
#include <hip/hip_runtime.h>
#include <hip/hip_bf16.h>

typedef _Float16 half16 __attribute__((ext_vector_type(16)));
typedef float    float8 __attribute__((ext_vector_type(8)));

#define NTOT   16384
#define CIN    128
#define HH     3

// LDS layout
#define WIN_H   49152            // 24 mtiles * 4 kchunks * 512 halves
#define WOUT_H  49152            // 8 otiles * 12 kchunks * 512 halves
#define PE_H    2048             // 4 kchunks * 512 halves
#define XS_F    (8 * 2048)       // per-wave x staging (f32)
#define SMEM_BYTES ((WIN_H + WOUT_H + PE_H) * 2 + (XS_F + 384 + 768 + 128 + 128) * 4)  // 271872

#if defined(__has_builtin)
#if __has_builtin(__builtin_amdgcn_global_load_async_to_lds_b128)
#define HAVE_ASYNC_LDS 1
#endif
#endif

typedef int vint4 __attribute__((vector_size(16)));
typedef __attribute__((address_space(1))) vint4 gvint4;   // global
typedef __attribute__((address_space(3))) vint4 lvint4;   // LDS

__device__ __forceinline__ float8 wmma16(half16 a, half16 b, float8 c) {
    return __builtin_amdgcn_wmma_f32_16x16x32_f16(false, a, false, b, (short)0, c, false, false);
}

__device__ __forceinline__ int kmap(int l, int i) {
    return ((l >> 4) << 3) + (i < 8 ? i : i + 8);
}

// 8 consecutive f32 from LDS as two b128 loads
__device__ __forceinline__ float8 ld8v(const float* p) {
    float4 a = *(const float4*)p;
    float4 b = *(const float4*)(p + 4);
    float8 r;
    r[0] = a.x; r[1] = a.y; r[2] = a.z; r[3] = a.w;
    r[4] = b.x; r[5] = b.y; r[6] = b.z; r[7] = b.w;
    return r;
}

// qk pair order: per head h -> q(4h), q(4h+2), k(12+4h), k(12+4h+2)
__device__ __forceinline__ constexpr int mt_of(int j) {
    int h = j >> 2, idx = j & 3;
    return (idx < 2) ? 4 * h + 2 * idx : 12 + 4 * h + 2 * (idx - 2);
}

struct PairBuf {           // one pipeline stage: 8 A-tiles + bias for both row groups
    half16 t[8];
    float8 bE, bO;
};

__device__ __forceinline__ void load_pair(PairBuf& B, const _Float16* wil,
                                          const float* bil, int mtE, int lane, int hoff) {
#pragma unroll
    for (int kk = 0; kk < 4; ++kk) {
        B.t[2 * kk]     = *(const half16*)(wil + (((mtE * 4) + kk) * 32 + lane) * 16);
        B.t[2 * kk + 1] = *(const half16*)(wil + ((((mtE + 1) * 4) + kk) * 32 + lane) * 16);
    }
    B.bE = ld8v(bil + mtE * 16 + hoff);
    B.bO = ld8v(bil + mtE * 16 + 16 + hoff);
}

__device__ __forceinline__ half16 pair_compute(const PairBuf& B, const half16* ybop) {
    float8 aE = B.bE, aO = B.bO;
#pragma unroll
    for (int kk = 0; kk < 4; ++kk) {
        aE = wmma16(B.t[2 * kk],     ybop[kk], aE);
        aO = wmma16(B.t[2 * kk + 1], ybop[kk], aO);
    }
    half16 out;
#pragma unroll
    for (int i = 0; i < 8; ++i) {
        out[i]     = (_Float16)aE[i];
        out[i + 8] = (_Float16)aO[i];
    }
    return out;
}

__global__ __launch_bounds__(256, 1)
void gtl_kernel(const float* __restrict__ x,
                const float* __restrict__ w_in,
                const float* __restrict__ b_in,
                const float* __restrict__ g_att,
                const float* __restrict__ w_out,
                const float* __restrict__ b_out,
                const float* __restrict__ bn_g,
                const float* __restrict__ bn_b,
                const float* __restrict__ bn_m,
                const float* __restrict__ bn_v,
                float* __restrict__ out) {
    extern __shared__ char smem[];
    _Float16* wil = (_Float16*)smem;
    _Float16* wol = wil + WIN_H;
    _Float16* pel = wol + WOUT_H;
    float* fbase  = (float*)(pel + PE_H);
    float* xsall  = fbase;                 // 8 * 2048
    float* bil    = xsall + XS_F;          // 384
    float* gal    = bil + 384;             // 768 : [h][lane][r] D-layout
    float* scl    = gal + 768;             // 128
    float* shf    = scl + 128;             // 128

    const int tid = threadIdx.x;

    // ---- stage w_in into f16 A-operand layout ----
    for (int o = tid; o < WIN_H; o += 256) {
        int i = o & 15, l = (o >> 4) & 31, t = o >> 9;   // t = mt*4+kc
        int kc = t & 3, mt = t >> 2;
        wil[o] = (_Float16)w_in[(16 * mt + (l & 15)) * CIN + 32 * kc + kmap(l, i)];
    }
    // ---- stage w_out into f16 A-operand layout ----
    for (int o = tid; o < WOUT_H; o += 256) {
        int i = o & 15, l = (o >> 4) & 31, t = o >> 9;   // t = ot*12+kcc
        int kcc = t % 12, ot = t / 12;
        wol[o] = (_Float16)w_out[(16 * ot + (l & 15)) * (HH * CIN) + 32 * kcc + kmap(l, i)];
    }
    // ---- positional encoding in B-operand chunk layout ----
    for (int o = tid; o < PE_H; o += 256) {
        int i = o & 15, l = (o >> 4) & 31, kc = o >> 9;
        int c = 32 * kc + kmap(l, i);
        float dv = __expf(-9.210340371976184f * (float)(c & ~1) * (1.0f / 128.0f));
        float ph = (float)(l & 15) * dv;
        pel[o] = (_Float16)((c & 1) ? __cosf(ph) : __sinf(ph));
    }
    // ---- small constants: b_in, global_att (D-layout), BN scale/shift ----
    for (int o = tid; o < 384; o += 256) bil[o] = b_in[o];
    for (int o = tid; o < 768; o += 256) {
        int r = o & 7, l = (o >> 3) & 31, h = o >> 8;
        gal[o] = g_att[h * 256 + (r + ((l >> 4) << 3)) * 16 + (l & 15)];
    }
    if (tid < 128) {
        float sc = bn_g[tid] * rsqrtf(bn_v[tid] + 1e-5f);
        scl[tid] = sc;
        shf[tid] = (b_out[tid] - bn_m[tid]) * sc + bn_b[tid];
    }
    __syncthreads();

    const int lane = tid & 31;
    const int wid  = tid >> 5;
    const int lo   = lane & 15;
    const int hoff = (lane >> 4) << 3;
    float* xs = xsall + wid * 2048;
    const float8 z8 = {0.f,0.f,0.f,0.f,0.f,0.f,0.f,0.f};
    const int nstride = gridDim.x * 8;

    for (int n = blockIdx.x * 8 + wid; n < NTOT; n += nstride) {
        const float* xn = x + (size_t)n * 2048;
        if (n + nstride < NTOT)
            __builtin_prefetch(x + (size_t)(n + nstride) * 2048 + lane * 16, 0, 1);

        // ---- 1. stage x[n] (16x128 f32) into LDS via async DMA ----
        asm volatile("s_wait_dscnt 0x0" ::: "memory");   // previous sample's DS reads
#ifdef HAVE_ASYNC_LDS
#pragma unroll
        for (int i = 0; i < 16; ++i) {
            __builtin_amdgcn_global_load_async_to_lds_b128(
                (gvint4*)(xn + (lane + 32 * i) * 4),
                (lvint4*)(xs + (lane + 32 * i) * 4),
                0, 0);
        }
        asm volatile("s_wait_asynccnt 0x0" ::: "memory");
#else
        {
            const float4* xn4 = (const float4*)xn;
            float4* xs4 = (float4*)xs;
#pragma unroll
            for (int i = 0; i < 16; ++i) xs4[lane + 32 * i] = xn4[lane + 32 * i];
            asm volatile("s_wait_dscnt 0x0" ::: "memory");
        }
#endif

        // ---- 2. y = x^T + pe as four 32x16 f16 B-operand chunks ----
        half16 ybop[4];
#pragma unroll
        for (int kc = 0; kc < 4; ++kc) {
            const float* p = xs + lo * 128 + 32 * kc + hoff;
            float8 l8 = ld8v(p);
            float8 h8 = ld8v(p + 16);
            half16 pv = *(const half16*)(pel + ((kc * 32 + lane) << 4));
            half16 yv;
#pragma unroll
            for (int j = 0; j < 8; ++j) {
                yv[j]     = (_Float16)l8[j];
                yv[j + 8] = (_Float16)h8[j];
            }
            ybop[kc] = yv + pv;
        }

        // ---- 3/4. q,k projections + attention: double-buffered pipeline over 12 pairs ----
        half16 attB[3];
        half16 qch[4];
        PairBuf pb[2];
        load_pair(pb[0], wil, bil, mt_of(0), lane, hoff);
#pragma unroll
        for (int j = 0; j < 12; ++j) {
            if (j < 11)
                load_pair(pb[(j + 1) & 1], wil, bil, mt_of(j + 1), lane, hoff);
            qch[j & 3] = pair_compute(pb[j & 1], ybop);

            if ((j & 3) == 3) {          // head complete -> attention + softmax
                int h = j >> 2;
                float8 aa = wmma16(qch[0], qch[2], z8);
                aa = wmma16(qch[1], qch[3], aa);
#pragma unroll
                for (int i = 0; i < 8; ++i) aa[i] *= (1.0f / 64.0f);

                float m = aa[0];
#pragma unroll
                for (int i = 1; i < 8; ++i) m = fmaxf(m, aa[i]);
                m = fmaxf(m, __shfl_xor(m, 16, 32));
                float s = 0.f;
#pragma unroll
                for (int i = 0; i < 8; ++i) { aa[i] = __expf(aa[i] - m); s += aa[i]; }
                s += __shfl_xor(s, 16, 32);
                float inv = 1.0f / s;

                float8 gv = ld8v(gal + (h * 32 + lane) * 8);
                half16 ab;
#pragma unroll
                for (int i = 0; i < 8; ++i) {
                    ab[i]     = (_Float16)(aa[i] * inv + gv[i]);
                    ab[i + 8] = (_Float16)0.0f;        // K padding (u >= 16)
                }
                attB[h] = ab;
            }
        }

        // ---- 5. xA tiles (x^T as f16 A operand, K=u padded to 32) ----
        half16 xA[8];
#pragma unroll
        for (int mt = 0; mt < 8; ++mt) {
            half16 v;
#pragma unroll
            for (int i = 0; i < 8; ++i) v[i] = (_Float16)xs[(hoff + i) * 128 + mt * 16 + lo];
#pragma unroll
            for (int i = 8; i < 16; ++i) v[i] = (_Float16)0.0f;
            xA[mt] = v;
        }

        float8 oacc[8];
#pragma unroll
        for (int ot = 0; ot < 8; ++ot) oacc[ot] = z8;

        // ---- y2 = x^T @ att fused into w_out GEMM: double-buffered wt tiles ----
        half16 wt[2][8];
#pragma unroll
        for (int ot = 0; ot < 8; ++ot)
            wt[0][ot] = *(const half16*)(wol + ((ot * 12 + 0) * 32 + lane) * 16);
#pragma unroll
        for (int j = 0; j < 12; ++j) {     // j == kcc == h*4 + kk
            if (j < 11) {
#pragma unroll
                for (int ot = 0; ot < 8; ++ot)
                    wt[(j + 1) & 1][ot] =
                        *(const half16*)(wol + ((ot * 12 + (j + 1)) * 32 + lane) * 16);
            }
            int kk = j & 3;
            float8 y0 = wmma16(xA[2 * kk],     attB[j >> 2], z8);
            float8 y1 = wmma16(xA[2 * kk + 1], attB[j >> 2], z8);
            half16 yB;
#pragma unroll
            for (int i = 0; i < 8; ++i) {
                yB[i]     = (_Float16)y0[i];
                yB[i + 8] = (_Float16)y1[i];
            }
#pragma unroll
            for (int ot = 0; ot < 8; ++ot)
                oacc[ot] = wmma16(wt[j & 1][ot], yB, oacc[ot]);
        }

        // ---- 6. BN + residual + ReLU, store transposed (n, v, c) ----
        float* op = out + (size_t)n * 2048 + lo * 128;
#pragma unroll
        for (int ot = 0; ot < 8; ++ot) {
            int c0 = ot * 16 + hoff;
            float8 sa = ld8v(scl + c0);
            float8 sh = ld8v(shf + c0);
            float8 xr = ld8v(xs + lo * 128 + c0);
#pragma unroll
            for (int r = 0; r < 8; ++r) {
                float val = xr[r] + oacc[ot][r] * sa[r] + sh[r];
                op[c0 + r] = fmaxf(val, 0.0f);
            }
        }
    }
}

extern "C" void kernel_launch(void* const* d_in, const int* in_sizes, int n_in,
                              void* d_out, int out_size, void* d_ws, size_t ws_size,
                              hipStream_t stream) {
    (void)in_sizes; (void)n_in; (void)out_size; (void)d_ws; (void)ws_size;
    const float* x     = (const float*)d_in[0];
    const float* w_in  = (const float*)d_in[1];
    const float* b_in  = (const float*)d_in[2];
    const float* g_att = (const float*)d_in[3];
    const float* w_out = (const float*)d_in[4];
    const float* b_out = (const float*)d_in[5];
    const float* bn_g  = (const float*)d_in[6];
    const float* bn_b  = (const float*)d_in[7];
    const float* bn_m  = (const float*)d_in[8];
    const float* bn_v  = (const float*)d_in[9];
    float* out = (float*)d_out;

    gtl_kernel<<<dim3(256), dim3(256), SMEM_BYTES, stream>>>(
        x, w_in, b_in, g_att, w_out, b_out, bn_g, bn_b, bn_m, bn_v, out);
}
